// GraphTransformerLayer_26190710571651
// MI455X (gfx1250) — compile-verified, matrix operand order checked by reading, not run on hardware
//
#include <hip/hip_runtime.h>
#include <hip/hip_bf16.h>

// ---------------------------------------------------------------------------
// GraphTransformer layer for MI455X (gfx1250, wave32, WMMA).
//
// Precision plan: f16 operands + f32 accumulation via v_wmma_f32_16x16x32_f16.
// Attention is flash-style (online softmax) so the 8x4096x4096 score tensor
// never touches HBM; adjacency A (64MB f32, read once per head) is the
// dominant memory term (~512MB -> ~22us at 23.3 TB/s) and is prefetched one
// key-tile ahead (global_prefetch_b8). Q/K/V/weights are L2 resident (192MB).
// All WMMA fragment loads are contiguous 16B loads:
//  - K rows serve directly as B-fragments of Q*K^T (contract dim d contiguous)
//  - V is pre-transposed (LDS-tiled, coalesced) to Vt[d][n]
//  - weights pre-transposed (LDS-tiled) to Wt[col][k]
// GEMMs are templated on K (512/1024) and fully unrolled; the attention tile
// loop keeps register pressure under the spill threshold by packing the
// adjacency mask into one bitmask up front and doing scale/mask/exp in place
// on the S accumulators (no separate probability array).
// BatchNorm is a coalesced 3-stage reduction.
// ---------------------------------------------------------------------------

typedef _Float16 f16;
typedef __attribute__((ext_vector_type(16))) _Float16 v16h;
typedef __attribute__((ext_vector_type(8)))  _Float16 v8h;
typedef __attribute__((ext_vector_type(8)))  float    v8f;

#define NNODES 4096
#define HID    512
#define HEADS  8
#define DH     64
#define FFN    1024

__device__ __forceinline__ v8f wmma_f16(v16h a, v16h b, v8f c) {
  // D = A(16x32 f16) * B(32x16 f16) + C(16x16 f32)
  return __builtin_amdgcn_wmma_f32_16x16x32_f16(false, a, false, b,
                                                (short)0, c, false, false);
}

// A/B fragment for 16x16x32 f16 WMMA.
// 16-bit A-matrix layout (ISA 7.12.2): lanes 0-15 hold row M=lane with
// K = {kb..kb+7, kb+16..kb+23}, kb = 0 for lanes 0-15, 8 for lanes 16-31.
// B mirrors this with N in place of M. `p` points at the start of the
// contract-dim-contiguous row/column (already offset by the 32-wide k-step).
__device__ __forceinline__ v16h load_frag(const f16* __restrict__ p, int kb) {
  v8h lo = *(const v8h*)(p + kb);
  v8h hi = *(const v8h*)(p + kb + 16);
  v16h r;
#pragma unroll
  for (int i = 0; i < 8; ++i) { r[i] = lo[i]; r[i + 8] = hi[i]; }
  return r;
}

// --------------------------- prep kernels ----------------------------------

__global__ void cvt_f32_to_f16(const float* __restrict__ x, f16* __restrict__ y,
                               int n) {
  int i = blockIdx.x * blockDim.x + threadIdx.x;
  if (i < n) y[i] = (f16)x[i];
}

// LDS-tiled transpose: dst[C][R] = (f16) src[R][C]. Coalesced on both sides
// (64x64 tile, padded LDS rows). Used for weight f32->f16t and V f16->f16t.
template <typename TI>
__global__ void transpose_tiled(const TI* __restrict__ src, f16* __restrict__ dst,
                                int R, int C) {
  __shared__ f16 tile[64][65];
  const int r0 = blockIdx.y * 64, c0 = blockIdx.x * 64;
  for (int k = threadIdx.x; k < 64 * 64; k += blockDim.x) {
    const int r = k >> 6, c = k & 63;
    tile[r][c] = (f16)src[(size_t)(r0 + r) * C + (c0 + c)];
  }
  __syncthreads();
  for (int k = threadIdx.x; k < 64 * 64; k += blockDim.x) {
    const int c = k >> 6, r = k & 63;   // consecutive k -> consecutive r -> coalesced
    dst[(size_t)(c0 + c) * R + (r0 + r)] = tile[r][c];
  }
}

// ------------------------------ WMMA GEMM ----------------------------------
// out[row][col] = A16[row][:] . Bt16[col][:] + bias[col] (+ residual) (relu?)
// Block: 128 threads = 4 waves; block tile 64 rows x 64 cols; each wave owns
// 16 rows x 64 cols (4 accumulators), so each A-fragment feeds 4 WMMAs.
// K is a compile-time constant (512 or 1024): the k-loop fully unrolls and
// the scheduler pipelines the load clauses across WMMAs on its own.
template <int K>
__global__ void gemm_wmma(const f16* __restrict__ A, int lda,
                          const f16* __restrict__ Bt,
                          const float* __restrict__ bias,
                          const float* __restrict__ residual,
                          float* __restrict__ outF, f16* __restrict__ outH,
                          int ldc, int relu) {
  const int lane   = threadIdx.x & 31;
  const int wave   = threadIdx.x >> 5;      // 0..3
  const int lanelo = lane & 15;
  const int kb     = (lane >> 4) ? 8 : 0;
  const int rowOff = (lane >> 4) * 8;
  const int colBase = blockIdx.x * 64;
  const int rowBase = blockIdx.y * 64 + wave * 16;

  const f16* Arow = A + (size_t)(rowBase + lanelo) * lda;
  const f16* Bcol[4];
#pragma unroll
  for (int c = 0; c < 4; ++c)
    Bcol[c] = Bt + (size_t)(colBase + c * 16 + lanelo) * K;

  v8f acc[4] = {};
#pragma unroll
  for (int k = 0; k < K; k += 32) {
    if ((k & 63) == 0 && k + 128 < K) {          // compile-time-resolved
      __builtin_prefetch(Arow + k + 128, 0, 0);
      __builtin_prefetch(Bcol[lane & 3] + k + 128, 0, 0);
    }
    v16h a = load_frag(Arow + k, kb);
#pragma unroll
    for (int c = 0; c < 4; ++c) {
      v16h b = load_frag(Bcol[c] + k, kb);
      acc[c] = wmma_f16(a, b, acc[c]);
    }
  }

#pragma unroll
  for (int c = 0; c < 4; ++c) {
    const int col = colBase + c * 16 + lanelo;
    const float bv = bias ? bias[col] : 0.f;
#pragma unroll
    for (int j = 0; j < 8; ++j) {
      const int row = rowBase + j + rowOff;     // C layout: M = j + 8*(lane>=16)
      const size_t idx = (size_t)row * ldc + col;
      float v = acc[c][j] + bv;
      if (residual) v += residual[idx];
      if (relu) v = fmaxf(v, 0.f);
      if (outF) outF[idx] = v;
      if (outH) outH[idx] = (f16)v;
    }
  }
}

// --------------------------- flash attention -------------------------------
// One workgroup = (head, 64 query rows); 4 waves, each owning 16 rows across
// ALL key columns -> online softmax is wave-private (16-lane shfl_xor trees,
// rows 0-7 live in lanes 0-15, rows 8-15 in lanes 16-31; masks 1/2/4/8 never
// cross the half). Per 64-key tile: 8 WMMAs for S, 8 WMMAs for O.
// Register-pressure discipline (the spill threshold is the real limit here):
//  - adjacency mask packed into ONE u32 bitmask before the S WMMAs, so the 32
//    global_load_b32 results recycle their dest VGPRs immediately
//  - scale/mask/exp applied IN PLACE on the S accumulators (no p[4][8] array)
__global__ void __launch_bounds__(128, 1)
attn_flash(const f16* __restrict__ Q, const f16* __restrict__ Kmat,
           const f16* __restrict__ Vt, const float* __restrict__ Adj,
           f16* __restrict__ O) {
  __shared__ __align__(16) f16 P[4][16][64];  // wave-private P tiles (8KB)

  const int lane   = threadIdx.x & 31;
  const int wave   = threadIdx.x >> 5;
  const int lanelo = lane & 15;
  const int kb     = (lane >> 4) ? 8 : 0;
  const int rowOff = (lane >> 4) * 8;
  const int head   = blockIdx.y;
  const int rowBase = blockIdx.x * 64 + wave * 16;

  // Q fragments for this wave's 16 rows: loaded once, resident all iterations.
  const f16* qrow = Q + (size_t)(rowBase + lanelo) * HID + head * DH;
  const v16h aq0 = load_frag(qrow, kb);        // d = 0..31
  const v16h aq1 = load_frag(qrow + 32, kb);   // d = 32..63

  float m[8], l[8];
  v8f o[4] = {};
#pragma unroll
  for (int j = 0; j < 8; ++j) { m[j] = -1e30f; l[j] = 0.f; }
  const float scale = 0.125f;                  // 1/sqrt(DH)

  for (int kt = 0; kt < NNODES; kt += 64) {
    // ---- prefetch next key tile (Adj rows: 16 rows x 2 lines per wave;
    //      K/Vt tiles are workgroup-uniform -> wave 0 only) ----
    if (kt + 64 < NNODES) {
      __builtin_prefetch(Adj + (size_t)(rowBase + lanelo) * NNODES +
                             (kt + 64) + (lane >> 4) * 32, 0, 0);
      if (wave == 0) {
        __builtin_prefetch(Kmat + (size_t)(kt + 64 + lane) * HID + head * DH, 0, 0);
        __builtin_prefetch(Kmat + (size_t)(kt + 96 + lane) * HID + head * DH, 0, 0);
        __builtin_prefetch(Vt + (size_t)(head * DH + lane) * NNODES + kt + 64, 0, 0);
        __builtin_prefetch(Vt + (size_t)(head * DH + 32 + lane) * NNODES + kt + 64, 0, 0);
      }
    }

    // ---- adjacency mask, packed: bit (c*8+j) = A[row][col] > 0.
    //      Issued before the WMMAs so load latency overlaps them and each
    //      loaded float is consumed immediately (dest VGPRs recycle). ----
    unsigned msk = 0;
#pragma unroll
    for (int c = 0; c < 4; ++c) {
      const float* acol = Adj + (size_t)(rowBase + rowOff) * NNODES +
                          kt + c * 16 + lanelo;
#pragma unroll
      for (int j = 0; j < 8; ++j)
        if (acol[(size_t)j * NNODES] > 0.f) msk |= 1u << (c * 8 + j);
    }

    // ---- S tile: 16 rows x 64 keys ----
    v8f s[4];
#pragma unroll
    for (int c = 0; c < 4; ++c) {
      const f16* krow = Kmat + (size_t)(kt + c * 16 + lanelo) * HID + head * DH;
      v16h b0 = load_frag(krow, kb);
      v16h b1 = load_frag(krow + 32, kb);
      v8f z = {};
      z = wmma_f16(aq0, b0, z);
      s[c] = wmma_f16(aq1, b1, z);
    }

    // ---- scale + mask in place (matches reference: scale then -1e9) ----
#pragma unroll
    for (int c = 0; c < 4; ++c)
#pragma unroll
      for (int j = 0; j < 8; ++j)
        s[c][j] = (msk & (1u << (c * 8 + j))) ? s[c][j] * scale : -1e9f;

    // ---- online softmax per owned row, in place on s ----
#pragma unroll
    for (int j = 0; j < 8; ++j) {
      float tmax = fmaxf(fmaxf(s[0][j], s[1][j]), fmaxf(s[2][j], s[3][j]));
#pragma unroll
      for (int d = 1; d < 16; d <<= 1) tmax = fmaxf(tmax, __shfl_xor(tmax, d, 32));
      const float mnew = fmaxf(m[j], tmax);
      const float corr = __expf(m[j] - mnew);
      float rsum = 0.f;
#pragma unroll
      for (int c = 0; c < 4; ++c) { s[c][j] = __expf(s[c][j] - mnew); rsum += s[c][j]; }
#pragma unroll
      for (int d = 1; d < 16; d <<= 1) rsum += __shfl_xor(rsum, d, 32);
      l[j] = l[j] * corr + rsum;
      m[j] = mnew;
#pragma unroll
      for (int c = 0; c < 4; ++c) o[c][j] *= corr;
    }

    // ---- relayout P: C-layout regs -> A-layout via wave-private LDS ----
#pragma unroll
    for (int c = 0; c < 4; ++c)
#pragma unroll
      for (int j = 0; j < 8; ++j)
        P[wave][j + rowOff][c * 16 + lanelo] = (f16)s[c][j];

    const f16* prow = &P[wave][lanelo][0];
    const v16h ap0 = load_frag(prow, kb);        // keys 0..31 of tile
    const v16h ap1 = load_frag(prow + 32, kb);   // keys 32..63

    // ---- O += P @ V  (B fragments contiguous from Vt[d][n]) ----
#pragma unroll
    for (int c = 0; c < 4; ++c) {
      const f16* vrow = Vt + (size_t)(head * DH + c * 16 + lanelo) * NNODES + kt;
      v16h b0 = load_frag(vrow, kb);
      v16h b1 = load_frag(vrow + 32, kb);
      o[c] = wmma_f16(ap0, b0, o[c]);
      o[c] = wmma_f16(ap1, b1, o[c]);
    }
  }

  // ---- normalize and emit O (f16, feeds the Wo GEMM) ----
#pragma unroll
  for (int c = 0; c < 4; ++c)
#pragma unroll
    for (int j = 0; j < 8; ++j) {
      const int row = rowBase + j + rowOff;
      const int col = head * DH + c * 16 + lanelo;
      O[(size_t)row * HID + col] = (f16)(o[c][j] / l[j]);
    }
}

// ------------------------- coalesced batchnorm -----------------------------
// Stage 1: per-(64-row-block, column) partial sum/sumsq; consecutive lanes hit
// consecutive columns -> fully coalesced. Deterministic (fixed partition).
__global__ void bn_partial(const float* __restrict__ x,
                           float* __restrict__ psum, float* __restrict__ psq,
                           int cols) {
  const int col = threadIdx.x;               // blockDim.x == cols == 512
  const int r0 = blockIdx.x * 64;
  float s = 0.f, q = 0.f;
  for (int r = r0; r < r0 + 64; ++r) {
    const float v = x[(size_t)r * cols + col];
    s += v; q += v * v;
  }
  psum[(size_t)blockIdx.x * cols + col] = s;
  psq [(size_t)blockIdx.x * cols + col] = q;
}

// Stage 2: combine partials, emit per-column scale/shift (y = x*sc + sh).
__global__ void bn_finalize(const float* __restrict__ psum,
                            const float* __restrict__ psq,
                            const float* __restrict__ g, const float* __restrict__ b,
                            float* __restrict__ scale, float* __restrict__ shift,
                            int nparts, int rows, int cols) {
  const int col = blockIdx.x * blockDim.x + threadIdx.x;
  if (col >= cols) return;
  float s = 0.f, q = 0.f;
  for (int p = 0; p < nparts; ++p) {
    s += psum[(size_t)p * cols + col];
    q += psq [(size_t)p * cols + col];
  }
  const float mean = s / rows;
  const float var  = q / rows - mean * mean;   // biased, matches jnp.var
  const float sc   = rsqrtf(var + 1e-5f) * g[col];
  scale[col] = sc;
  shift[col] = b[col] - mean * sc;
}

// Stage 3: coalesced elementwise apply (+ optional f16 copy for next GEMM).
__global__ void bn_apply(const float* __restrict__ x,
                         const float* __restrict__ scale,
                         const float* __restrict__ shift,
                         float* __restrict__ y, f16* __restrict__ y16,
                         int n, int cols) {
  const int i = blockIdx.x * blockDim.x + threadIdx.x;
  if (i >= n) return;
  const int col = i % cols;                    // cols is a power of two
  const float v = x[i] * scale[col] + shift[col];
  y[i] = v;
  if (y16) y16[i] = (f16)v;
}

// ------------------------------ launcher -----------------------------------

extern "C" void kernel_launch(void* const* d_in, const int* in_sizes, int n_in,
                              void* d_out, int out_size, void* d_ws, size_t ws_size,
                              hipStream_t stream) {
  (void)in_sizes; (void)n_in; (void)out_size; (void)ws_size;
  const float* A    = (const float*)d_in[0];
  const float* h    = (const float*)d_in[1];
  const float* Wq   = (const float*)d_in[2];  const float* bq = (const float*)d_in[3];
  const float* Wk   = (const float*)d_in[4];  const float* bk = (const float*)d_in[5];
  const float* Wv   = (const float*)d_in[6];  const float* bv = (const float*)d_in[7];
  const float* Wo   = (const float*)d_in[8];  const float* bo = (const float*)d_in[9];
  const float* bn1g = (const float*)d_in[10]; const float* bn1b = (const float*)d_in[11];
  const float* bn2g = (const float*)d_in[12]; const float* bn2b = (const float*)d_in[13];
  const float* W1   = (const float*)d_in[14]; const float* b1 = (const float*)d_in[15];
  const float* W2   = (const float*)d_in[16]; const float* b2 = (const float*)d_in[17];
  float* out = (float*)d_out;

  // ---- workspace carve (~64 MiB) ----
  char* w = (char*)d_ws;
  auto alloc = [&](size_t bytes) { char* p = w; w += (bytes + 255) & ~size_t(255); return p; };
  f16*   h16   = (f16*)  alloc((size_t)NNODES * HID * 2);
  f16*   Wqt   = (f16*)  alloc((size_t)HID * HID * 2);
  f16*   Wkt   = (f16*)  alloc((size_t)HID * HID * 2);
  f16*   Wvt   = (f16*)  alloc((size_t)HID * HID * 2);
  f16*   Wot   = (f16*)  alloc((size_t)HID * HID * 2);
  f16*   W1t   = (f16*)  alloc((size_t)HID * FFN * 2);
  f16*   W2t   = (f16*)  alloc((size_t)FFN * HID * 2);
  f16*   Q16   = (f16*)  alloc((size_t)NNODES * HID * 2);
  f16*   K16   = (f16*)  alloc((size_t)NNODES * HID * 2);
  f16*   V16   = (f16*)  alloc((size_t)NNODES * HID * 2);
  f16*   Vt16  = (f16*)  alloc((size_t)NNODES * HID * 2);
  f16*   O16   = (f16*)  alloc((size_t)NNODES * HID * 2);
  float* x1    = (float*)alloc((size_t)NNODES * HID * 4);
  float* hmid  = (float*)alloc((size_t)NNODES * HID * 4);
  f16*   hmid16= (f16*)  alloc((size_t)NNODES * HID * 2);
  f16*   y1    = (f16*)  alloc((size_t)NNODES * FFN * 2);
  float* x2    = (float*)alloc((size_t)NNODES * HID * 4);
  float* psum  = (float*)alloc((size_t)64 * HID * 4);
  float* psq   = (float*)alloc((size_t)64 * HID * 4);
  float* bsc   = (float*)alloc((size_t)HID * 4);
  float* bsh   = (float*)alloc((size_t)HID * 4);

  const int NE = NNODES * HID;   // 2M elements

  // ---- precision/layout prep (tiled, coalesced) ----
  cvt_f32_to_f16<<<(NE + 255) / 256, 256, 0, stream>>>(h, h16, NE);
  transpose_tiled<float><<<dim3(HID / 64, HID / 64), 256, 0, stream>>>(Wq, Wqt, HID, HID);
  transpose_tiled<float><<<dim3(HID / 64, HID / 64), 256, 0, stream>>>(Wk, Wkt, HID, HID);
  transpose_tiled<float><<<dim3(HID / 64, HID / 64), 256, 0, stream>>>(Wv, Wvt, HID, HID);
  transpose_tiled<float><<<dim3(HID / 64, HID / 64), 256, 0, stream>>>(Wo, Wot, HID, HID);
  transpose_tiled<float><<<dim3(FFN / 64, HID / 64), 256, 0, stream>>>(W1, W1t, HID, FFN);
  transpose_tiled<float><<<dim3(HID / 64, FFN / 64), 256, 0, stream>>>(W2, W2t, FFN, HID);

  // ---- Q/K/V projections (f16 out) ----
  dim3 gHID(HID / 64, NNODES / 64);
  gemm_wmma<HID><<<gHID, 128, 0, stream>>>(h16, HID, Wqt, bq, nullptr, nullptr, Q16, HID, 0);
  gemm_wmma<HID><<<gHID, 128, 0, stream>>>(h16, HID, Wkt, bk, nullptr, nullptr, K16, HID, 0);
  gemm_wmma<HID><<<gHID, 128, 0, stream>>>(h16, HID, Wvt, bv, nullptr, nullptr, V16, HID, 0);
  transpose_tiled<f16><<<dim3(HID / 64, NNODES / 64), 256, 0, stream>>>(V16, Vt16, NNODES, HID);

  // ---- masked multi-head attention (flash, online softmax) ----
  attn_flash<<<dim3(NNODES / 64, HEADS), 128, 0, stream>>>(Q16, K16, Vt16, A, O16);

  // ---- output projection + residual, BN1 ----
  gemm_wmma<HID><<<gHID, 128, 0, stream>>>(O16, HID, Wot, bo, h, x1, nullptr, HID, 0);
  bn_partial<<<NNODES / 64, HID, 0, stream>>>(x1, psum, psq, HID);
  bn_finalize<<<(HID + 255) / 256, 256, 0, stream>>>(psum, psq, bn1g, bn1b, bsc, bsh,
                                                     NNODES / 64, NNODES, HID);
  bn_apply<<<(NE + 255) / 256, 256, 0, stream>>>(x1, bsc, bsh, hmid, hmid16, NE, HID);

  // ---- FFN: relu(hmid @ W1 + b1) @ W2 + b2 + residual, BN2 ----
  gemm_wmma<HID><<<dim3(FFN / 64, NNODES / 64), 128, 0, stream>>>(
      hmid16, HID, W1t, b1, nullptr, nullptr, y1, FFN, 1);
  gemm_wmma<FFN><<<gHID, 128, 0, stream>>>(y1, FFN, W2t, b2, hmid, x2, nullptr, HID, 0);
  bn_partial<<<NNODES / 64, HID, 0, stream>>>(x2, psum, psq, HID);
  bn_finalize<<<(HID + 255) / 256, 256, 0, stream>>>(psum, psq, bn2g, bn2b, bsc, bsh,
                                                     NNODES / 64, NNODES, HID);
  bn_apply<<<(NE + 255) / 256, 256, 0, stream>>>(x2, bsc, bsh, out, nullptr, NE, HID);
}
